// Head_27650999452139
// MI455X (gfx1250) — matrix operand order repaired and checked by reading, not compiled, not run on hardware
//
#include <hip/hip_runtime.h>
#include <hip/hip_bf16.h>

// Problem constants (match reference)
#define BB 128
#define TT 512
#define CC 256
#define HH 64

typedef __attribute__((ext_vector_type(16))) _Float16 v16h;
typedef __attribute__((ext_vector_type(8)))  _Float16 v8h;
typedef __attribute__((ext_vector_type(8)))  float    v8f;
typedef __attribute__((ext_vector_type(4)))  float    v4f;

static __device__ __forceinline__ v16h concat8(v8h lo, v8h hi) {
  return __builtin_shufflevector(lo, hi, 0,1,2,3,4,5,6,7,8,9,10,11,12,13,14,15);
}

static __device__ __forceinline__ float half16_max(float v) {
  v = fmaxf(v, __shfl_xor(v, 1, 32));
  v = fmaxf(v, __shfl_xor(v, 2, 32));
  v = fmaxf(v, __shfl_xor(v, 4, 32));
  v = fmaxf(v, __shfl_xor(v, 8, 32));
  return v;  // reduced across the 16-lane half (C-layout rows live in one half)
}

static __device__ __forceinline__ float half16_sum(float v) {
  v += __shfl_xor(v, 1, 32);
  v += __shfl_xor(v, 2, 32);
  v += __shfl_xor(v, 4, 32);
  v += __shfl_xor(v, 8, 32);
  return v;
}

// ---------------------------------------------------------------------------
// Kernel 0: pack Wq/Wk/Wv [C,H] fp32 into exact WMMA B-operand layout, f16.
// wpack[w][kstep(8)][ntile(4)][lane(32)][16 halfs]
//   lane: n = ntile*16 + (lane&15); k = kstep*32 + 16*(lane>>4) + j
// Then a B fragment in the GEMM is ONE contiguous 32B load per lane.
// ---------------------------------------------------------------------------
__global__ __launch_bounds__(32)
void pack_w_kernel(const float* __restrict__ Wq, const float* __restrict__ Wk,
                   const float* __restrict__ Wv, _Float16* __restrict__ wpack) {
  const int lane  = threadIdx.x;            // 0..31
  const int idx   = blockIdx.x;             // 0..95  = w*32 + kstep*4 + ntile
  const int w     = idx >> 5;
  const int kstep = (idx & 31) >> 2;
  const int nt    = idx & 3;
  const float* W  = (w == 0) ? Wq : (w == 1) ? Wk : Wv;
  const int n     = nt * 16 + (lane & 15);
  const int kbase = kstep * 32 + 16 * (lane >> 4);
  _Float16* dst   = wpack + ((((w * 8 + kstep) * 4 + nt) * 32 + lane) * 16);
#pragma unroll
  for (int j = 0; j < 16; ++j) dst[j] = (_Float16)W[(size_t)(kbase + j) * HH + n];
}

// ---------------------------------------------------------------------------
// Kernel 1: QKV projection.  One wave per (b, 16-row tile of T).
//   [16 x 64] = x[16 x 256] * W[256 x 64], 8 k-steps of v_wmma_f32_16x16x32_f16
// Outputs: Qh,Kh as f16 [B][T][H];  V stored transposed Vt f16 [B][H][T].
// ---------------------------------------------------------------------------
__global__ __launch_bounds__(32)
void qkv_proj_kernel(const float* __restrict__ x, const _Float16* __restrict__ wpack,
                     _Float16* __restrict__ Qh, _Float16* __restrict__ Kh,
                     _Float16* __restrict__ Vt) {
  const int lane  = threadIdx.x;
  const int b     = blockIdx.x >> 5;        // T/16 = 32 tiles per batch
  const int ttile = blockIdx.x & 31;
  const int col   = lane & 15;
  const int half  = lane >> 4;

  const float* xrow = x + (((size_t)b * TT) + ttile * 16 + col) * CC;

  v8f acc[3][4] = {};
#pragma unroll
  for (int kstep = 0; kstep < 8; ++kstep) {
    // A fragment (16x32 f16): lane holds row=col; 16 contiguous K:
    //   K=[k0..k0+16), k0 = kstep*32 + 16*half
    //   (matches wpack B layout)
    const int k0 = kstep * 32 + half * 16;
    const v4f c0 = *(const v4f*)(xrow + k0);
    const v4f c1 = *(const v4f*)(xrow + k0 + 4);
    const v4f c2 = *(const v4f*)(xrow + k0 + 8);
    const v4f c3 = *(const v4f*)(xrow + k0 + 12);
    v16h a;
#pragma unroll
    for (int j = 0; j < 4; ++j) {
      a[j]      = (_Float16)c0[j];
      a[4 + j]  = (_Float16)c1[j];
      a[8 + j]  = (_Float16)c2[j];
      a[12 + j] = (_Float16)c3[j];
    }
#pragma unroll
    for (int w = 0; w < 3; ++w) {
#pragma unroll
      for (int nt = 0; nt < 4; ++nt) {
        const v16h bfrag =
            *(const v16h*)(wpack + ((((w * 8 + kstep) * 4 + nt) * 32 + lane) * 16));
        acc[w][nt] = __builtin_amdgcn_wmma_f32_16x16x32_f16(
            false, a, false, bfrag, (short)0, acc[w][nt], false, false);
      }
    }
  }

  // C layout: VGPR r, lane -> element (m = r + 8*half, n = nt*16 + col)
#pragma unroll
  for (int nt = 0; nt < 4; ++nt) {
#pragma unroll
    for (int r = 0; r < 8; ++r) {
      const int m  = r + 8 * half;
      const int tq = ttile * 16 + m;
      const int n  = nt * 16 + col;
      const size_t qk = (((size_t)b * TT) + tq) * HH + n;
      Qh[qk] = (_Float16)acc[0][nt][r];
      Kh[qk] = (_Float16)acc[1][nt][r];
      Vt[((size_t)b * HH + n) * TT + tq] = (_Float16)acc[2][nt][r];  // transposed
    }
  }
}

// ---------------------------------------------------------------------------
// Kernel 2: fused causal attention (flash-style, online softmax).
// One wave per (b, 16-query tile). Iterate 32-key pairs up to the diagonal.
// Next-iteration K/V tiles are prefetched (global_prefetch_b8) so the
// softmax/LDS phase overlaps the following tile's memory fetch.
// ---------------------------------------------------------------------------
__global__ __launch_bounds__(32)
void attn_kernel(const _Float16* __restrict__ Qh, const _Float16* __restrict__ Kh,
                 const _Float16* __restrict__ Vt, float* __restrict__ out) {
  __shared__ __align__(16) _Float16 Plds[16][40];  // 16 q-rows x 32 keys, +8 pad halfs

  const int lane = threadIdx.x;
  const int b    = blockIdx.x >> 5;
  const int qt   = blockIdx.x & 31;
  const int col  = lane & 15;
  const int half = lane >> 4;

  const _Float16* Kbase = Kh + ((size_t)b * TT) * HH;            // [T][H]
  const _Float16* Vlane = Vt + ((size_t)b * HH + col) * TT;      // lane's V column
  // This lane's K fragment base: key row = col within a tile, hdim chunk by half
  const _Float16* Klane = Kbase + (size_t)col * HH + 16 * half;

  // Q A-fragments over head dim H=64 (2 k-steps of 32)
  v16h Aq[2];
  {
    const _Float16* qrow = Qh + (((size_t)b * TT) + qt * 16 + col) * HH;
#pragma unroll
    for (int ks = 0; ks < 2; ++ks) {
      const int k0 = ks * 32 + half * 16;
      const v8h lo = *(const v8h*)(qrow + k0);
      const v8h hi = *(const v8h*)(qrow + k0 + 8);
      Aq[ks] = concat8(lo, hi);
    }
  }

  v8f  O[4] = {};
  float mrow[8], lrow[8];
#pragma unroll
  for (int r = 0; r < 8; ++r) { mrow[r] = -1e30f; lrow[r] = 0.f; }

  const int npairs = (qt >> 1) + 1;
  for (int pt = 0; pt < npairs; ++pt) {
    // ---- prefetch next pair's K and V tiles (global_prefetch_b8) ----
    if (pt + 1 < npairs) {
      const int np = pt + 1;
      __builtin_prefetch(Klane + (size_t)(2 * np) * 16 * HH, 0, 3);
      __builtin_prefetch(Klane + (size_t)(2 * np + 1) * 16 * HH, 0, 3);
      __builtin_prefetch(Vlane + np * 32 + 16 * half, 0, 3);
      __builtin_prefetch(Vlane + (size_t)48 * TT + np * 32 + 16 * half, 0, 3);
    }

    // ---- S = (Q K^T) * 1/sqrt(H) for two 16-key tiles, causal-masked ----
    v8f S[2];
#pragma unroll
    for (int t = 0; t < 2; ++t) {
      const int st = 2 * pt + t;  // key tile index, always <= 31
      // B fragment: lane holds key = st*16+col, hdims k = ks*32+16*half+[0..16)
      v8f s = {};
#pragma unroll
      for (int ks = 0; ks < 2; ++ks) {
        const _Float16* krow = Klane + (size_t)(st * 16) * HH + ks * 32;
        const v16h bk = concat8(*(const v8h*)krow, *(const v8h*)(krow + 8));
        s = __builtin_amdgcn_wmma_f32_16x16x32_f16(
            false, Aq[ks], false, bk, (short)0, s, false, false);
      }
      const int s_glob = st * 16 + col;  // key column owned by this lane
#pragma unroll
      for (int r = 0; r < 8; ++r) {
        const int q_glob = qt * 16 + r + 8 * half;
        float v = s[r] * 0.125f;                      // 1/sqrt(64)
        s[r] = (s_glob > q_glob) ? -1e30f : v;        // causal mask
      }
      S[t] = s;
    }

    // ---- online softmax statistics (row reductions across 16-lane half) ----
    float newm[8], alpha[8];
#pragma unroll
    for (int r = 0; r < 8; ++r) {
      const float mt = half16_max(fmaxf(S[0][r], S[1][r]));
      newm[r]  = fmaxf(mrow[r], mt);
      alpha[r] = __expf(mrow[r] - newm[r]);
      mrow[r]  = newm[r];
      const float p0 = __expf(S[0][r] - newm[r]);
      const float p1 = __expf(S[1][r] - newm[r]);
      S[0][r] = p0; S[1][r] = p1;
      const float rs = half16_sum(p0 + p1);
      lrow[r] = lrow[r] * alpha[r] + rs;
    }

    // ---- C-layout -> A-layout transpose of P through LDS (f16) ----
#pragma unroll
    for (int r = 0; r < 8; ++r) {
      const int m = r + 8 * half;
      Plds[m][col]      = (_Float16)S[0][r];
      Plds[m][16 + col] = (_Float16)S[1][r];
    }
    // same-wave DS ops are in-order; compiler inserts s_wait_dscnt before use
    const v8h plo = *(const v8h*)&Plds[col][half * 16];
    const v8h phi = *(const v8h*)&Plds[col][half * 16 + 8];
    const v16h Ap = concat8(plo, phi);

    // ---- O = O*alpha + P @ V  (V transposed: contiguous B fragments) ----
#pragma unroll
    for (int nt = 0; nt < 4; ++nt) {
#pragma unroll
      for (int r = 0; r < 8; ++r) O[nt][r] *= alpha[r];
      const _Float16* vp = Vlane + (size_t)(nt * 16) * TT + pt * 32 + 16 * half;
      const v16h Bv = concat8(*(const v8h*)vp, *(const v8h*)(vp + 8));
      O[nt] = __builtin_amdgcn_wmma_f32_16x16x32_f16(
          false, Ap, false, Bv, (short)0, O[nt], false, false);
    }
  }

  // ---- normalize and store fp32 output [B][T][H] ----
#pragma unroll
  for (int nt = 0; nt < 4; ++nt) {
#pragma unroll
    for (int r = 0; r < 8; ++r) {
      const int m = r + 8 * half;
      const size_t o = (((size_t)b * TT) + qt * 16 + m) * HH + nt * 16 + col;
      out[o] = O[nt][r] / lrow[r];
    }
  }
}

// ---------------------------------------------------------------------------
extern "C" void kernel_launch(void* const* d_in, const int* in_sizes, int n_in,
                              void* d_out, int out_size, void* d_ws, size_t ws_size,
                              hipStream_t stream) {
  const float* x  = (const float*)d_in[0];
  const float* Wq = (const float*)d_in[1];
  const float* Wk = (const float*)d_in[2];
  const float* Wv = (const float*)d_in[3];
  float* out = (float*)d_out;

  // Workspace layout: [wpack 96KB][pad to 128KB][Qh 8MB][Kh 8MB][Vt 8MB]
  _Float16* wpack = (_Float16*)d_ws;
  _Float16* Qh = (_Float16*)((char*)d_ws + (size_t)(128 << 10));
  _Float16* Kh = Qh + (size_t)BB * TT * HH;
  _Float16* Vt = Kh + (size_t)BB * TT * HH;

  pack_w_kernel<<<96, 32, 0, stream>>>(Wq, Wk, Wv, wpack);
  qkv_proj_kernel<<<BB * (TT / 16), 32, 0, stream>>>(x, wpack, Qh, Kh, Vt);
  attn_kernel<<<BB * (TT / 16), 32, 0, stream>>>(Qh, Kh, Vt, out);
}
